// HNO_35072702939553
// MI455X (gfx1250) — matrix operand order repaired and checked by reading, compiled, not run on hardware
//
#include <hip/hip_runtime.h>

// ---------------------------------------------------------------------------
// ChebConv GNN pipeline for gfx1250 (MI455X).
//   N=50000 nodes, E=800000 edges, H=128, K=4 (sizes taken from in_sizes).
// Dense GEMMs (layers 2-4: [N,512] @ [512,128]) use v_wmma_f32_16x16x32_bf16.
// Sparse props use L2-resident global_atomic_add_f32 scatter.
// ---------------------------------------------------------------------------

typedef __attribute__((ext_vector_type(16))) __bf16 v16bf;
typedef __attribute__((ext_vector_type(8)))  float  v8f;

union FragU { uint4 u4[2]; v16bf v; };

__device__ __forceinline__ unsigned short f2bf(float f) {
    unsigned int u = __float_as_uint(f);
    unsigned int r = u + 0x7FFFu + ((u >> 16) & 1u);   // round-to-nearest-even
    return (unsigned short)(r >> 16);
}

// ------------------------------ utility kernels ----------------------------

__global__ void k_zero_f32(float* __restrict__ p, int n) {
    int i = blockIdx.x * blockDim.x + threadIdx.x;
    for (; i < n; i += gridDim.x * blockDim.x) p[i] = 0.0f;
}

// deg[src[e]] += 1
__global__ void k_degree(const int* __restrict__ src, float* __restrict__ deg, int E) {
    int e = blockIdx.x * blockDim.x + threadIdx.x;
    if (e < E) atomicAdd(&deg[src[e]], 1.0f);
}

// w[e] = -dinv(src)*dinv(dst)
__global__ void k_edgew(const int* __restrict__ src, const int* __restrict__ dst,
                        const float* __restrict__ deg, float* __restrict__ w, int E) {
    int e = blockIdx.x * blockDim.x + threadIdx.x;
    if (e >= E) return;
    float ds = deg[src[e]], dd = deg[dst[e]];
    float a = ds > 0.0f ? rsqrtf(fmaxf(ds, 1.0f)) : 0.0f;
    float b = dd > 0.0f ? rsqrtf(fmaxf(dd, 1.0f)) : 0.0f;
    w[e] = -a * b;
}

// ------------------------------ sparse props -------------------------------

// out[dst] += in[src]*w  for F=128; one wave per edge, 4 floats per lane.
__global__ void k_prop128(const float* __restrict__ in, const int* __restrict__ src,
                          const int* __restrict__ dst, const float* __restrict__ w,
                          float* __restrict__ out, int E) {
    int lane = threadIdx.x & 31;
    int warp = threadIdx.x >> 5;
    int e = blockIdx.x * 8 + warp;
    if (e >= E) return;
    int s = src[e], d = dst[e];
    float we = w[e];
    const float4* ip = (const float4*)(in + (size_t)s * 128);
    float*        op = out + (size_t)d * 128;
    float4 v = ip[lane];
    atomicAdd(op + lane * 4 + 0, v.x * we);
    atomicAdd(op + lane * 4 + 1, v.y * we);
    atomicAdd(op + lane * 4 + 2, v.z * we);
    atomicAdd(op + lane * 4 + 3, v.w * we);
}

// F=3 variant: one thread per edge
__global__ void k_prop3(const float* __restrict__ in, const int* __restrict__ src,
                        const int* __restrict__ dst, const float* __restrict__ w,
                        float* __restrict__ out, int E) {
    int e = blockIdx.x * blockDim.x + threadIdx.x;
    if (e >= E) return;
    int s = src[e], d = dst[e];
    float we = w[e];
    const float* ip = in + (size_t)s * 3;
    float*       op = out + (size_t)d * 3;
    atomicAdd(op + 0, ip[0] * we);
    atomicAdd(op + 1, ip[1] * we);
    atomicAdd(op + 2, ip[2] * we);
}

// out[i] = 2*t[i] - a[i]   (Chebyshev recursion; out may alias a)
__global__ void k_cheb_next(const float* __restrict__ t, const float* __restrict__ a,
                            float* __restrict__ out, int n) {
    int i = blockIdx.x * blockDim.x + threadIdx.x;
    if (i < n) out[i] = 2.0f * t[i] - a[i];
}

// convert f32 [N,128] into bf16 column block of Abf [N,512]
__global__ void k_cvt_bf16(const float* __restrict__ in, unsigned short* __restrict__ Abf,
                           int colOff, int n128) {
    int i = blockIdx.x * blockDim.x + threadIdx.x;
    if (i >= n128) return;
    int row = i >> 7, c = i & 127;
    Abf[(size_t)row * 512 + colOff + c] = f2bf(in[i]);
}

// ------------------------------ weight packing -----------------------------
// Pack f32 W [4][128][128] into WMMA-fragment-native bf16 layout:
//   Bp[(kt*8 + ct)*512 + lane*16 + e] = bf16( W[k>>7][k&127][ct*16 + (lane&15)] )
//   with k = kt*32 + (lane>>4)*16 + e.
__global__ void k_pack_w(const float* __restrict__ W, unsigned short* __restrict__ Bp) {
    int i = blockIdx.x * blockDim.x + threadIdx.x;   // 65536 total
    int e    = i & 15;
    int lane = (i >> 4) & 31;
    int ct   = (i >> 9) & 7;
    int kt   = i >> 12;
    int k    = kt * 32 + (lane >> 4) * 16 + e;       // 0..511
    int col  = ct * 16 + (lane & 15);
    Bp[i] = f2bf(W[(size_t)k * 128 + col]);          // W viewed as [512][128]
}

// ------------------------------ WMMA GEMM ----------------------------------
// C[N,128] = act( A[N,512](bf16) @ Bp(packed bf16) + bias )
// block = 256 threads = 8 waves; wave ct owns cols [16ct,16ct+16); block owns 16 rows.
// act: 0 = none, 1 = leaky_relu(0.01), 2 = relu
__global__ void k_gemm_wmma(const unsigned short* __restrict__ A,
                            const unsigned short* __restrict__ Bp,
                            const float* __restrict__ bias,
                            float* __restrict__ C, int nrows, int act) {
    const int lane = threadIdx.x & 31;
    const int ct   = threadIdx.x >> 5;          // 0..7 column tile
    const int m    = lane & 15;
    const int half = lane >> 4;
    const int rowBase = blockIdx.x * 16;

    int ar = rowBase + m; if (ar > nrows - 1) ar = nrows - 1;   // clamp for safe loads
    const unsigned short* arow = A + (size_t)ar * 512;

    v8f acc = {};
#pragma unroll
    for (int kt = 0; kt < 16; ++kt) {
        const int k0 = kt * 32 + half * 16;
        FragU a, b;
        a.u4[0] = *(const uint4*)(arow + k0);
        a.u4[1] = *(const uint4*)(arow + k0 + 8);
        const unsigned short* bblk = Bp + ((kt * 8 + ct) * 512 + lane * 16);
        b.u4[0] = *(const uint4*)(bblk);
        b.u4[1] = *(const uint4*)(bblk + 8);
        if (kt + 1 < 16)
            __builtin_prefetch(arow + k0 + 32, 0, 1);   // global_prefetch_b8 on A stream
        acc = __builtin_amdgcn_wmma_f32_16x16x32_bf16(
            /*neg_a=*/false, a.v, /*neg_b=*/false, b.v,
            /*c_mod=*/(short)0, acc, /*reuse_a=*/false, /*reuse_b=*/false);
    }

    const int col = ct * 16 + m;
    const float bv = bias[col];
#pragma unroll
    for (int r = 0; r < 8; ++r) {
        int row = rowBase + half * 8 + r;
        if (row >= nrows) continue;
        float v = acc[r] + bv;
        if (act == 1) v = v > 0.0f ? v : 0.01f * v;
        else if (act == 2) v = fmaxf(v, 0.0f);
        C[(size_t)row * 128 + col] = v;
    }
}

// ------------------------------ layer 1 (K-dim = 12) -----------------------

__global__ void k_gemm_l1(const float* __restrict__ x, const float* __restrict__ c1,
                          const float* __restrict__ c2, const float* __restrict__ c3,
                          const float* __restrict__ W1, const float* __restrict__ b1,
                          float* __restrict__ out, int n128) {
    int i = blockIdx.x * blockDim.x + threadIdx.x;
    if (i >= n128) return;
    int n = i >> 7, c = i & 127;
    float acc = b1[c];
    const float* bases[4] = { x, c1, c2, c3 };
#pragma unroll
    for (int k = 0; k < 4; ++k) {
        const float* xv = bases[k] + (size_t)n * 3;
        const float* Wk = W1 + (size_t)k * 3 * 128;
        acc += xv[0] * Wk[0 * 128 + c] + xv[1] * Wk[1 * 128 + c] + xv[2] * Wk[2 * 128 + c];
    }
    acc = acc > 0.0f ? acc : 0.01f * acc;   // leaky_relu
    out[i] = acc;
}

// ------------------------------ BatchNorm ----------------------------------

__global__ void k_bn_stats(const float* __restrict__ h, float* __restrict__ mean,
                           float* __restrict__ istd, int N) {
    const int c = blockIdx.x;   // 0..127
    float s = 0.0f, s2 = 0.0f;
    for (int r = threadIdx.x; r < N; r += blockDim.x) {
        float v = h[(size_t)r * 128 + c];
        s += v; s2 += v * v;
    }
    __shared__ float bs[256], bq[256];
    bs[threadIdx.x] = s; bq[threadIdx.x] = s2;
    __syncthreads();
    for (int st = 128; st > 0; st >>= 1) {
        if (threadIdx.x < st) { bs[threadIdx.x] += bs[threadIdx.x + st];
                                bq[threadIdx.x] += bq[threadIdx.x + st]; }
        __syncthreads();
    }
    if (threadIdx.x == 0) {
        float m  = bs[0] / (float)N;
        float vr = bq[0] / (float)N - m * m;
        mean[c] = m;
        istd[c] = rsqrtf(vr + 1e-5f);
    }
}

__global__ void k_bn_apply(const float* __restrict__ in, const float* __restrict__ mean,
                           const float* __restrict__ istd, const float* __restrict__ g,
                           const float* __restrict__ be, float* __restrict__ out, int n128) {
    int i = blockIdx.x * blockDim.x + threadIdx.x;
    if (i >= n128) return;
    int c = i & 127;
    out[i] = (in[i] - mean[c]) * istd[c] * g[c] + be[c];
}

// ------------------------------ final L2 normalize -------------------------

__global__ void k_l2norm(const float* __restrict__ in, float* __restrict__ out) {
    const int row = blockIdx.x;
    const int t   = threadIdx.x;    // 128 threads
    float v = in[(size_t)row * 128 + t];
    __shared__ float sh[128];
    sh[t] = v * v;
    __syncthreads();
    for (int s = 64; s > 0; s >>= 1) {
        if (t < s) sh[t] += sh[t + s];
        __syncthreads();
    }
    float nrm = sqrtf(sh[0]);
    out[(size_t)row * 128 + t] = v / fmaxf(nrm, 1e-12f);
}

// ---------------------------------------------------------------------------

extern "C" void kernel_launch(void* const* d_in, const int* in_sizes, int n_in,
                              void* d_out, int out_size, void* d_ws, size_t ws_size,
                              hipStream_t stream) {
    const float* x  = (const float*)d_in[0];
    const int*   ei = (const int*)d_in[1];
    const float* W1 = (const float*)d_in[2];  const float* b1 = (const float*)d_in[3];
    const float* W2 = (const float*)d_in[4];  const float* b2 = (const float*)d_in[5];
    const float* W3 = (const float*)d_in[6];  const float* b3 = (const float*)d_in[7];
    const float* W4 = (const float*)d_in[8];  const float* b4 = (const float*)d_in[9];
    const float* g1 = (const float*)d_in[10]; const float* be1 = (const float*)d_in[11];
    const float* g2 = (const float*)d_in[12]; const float* be2 = (const float*)d_in[13];
    const float* g3 = (const float*)d_in[14]; const float* be3 = (const float*)d_in[15];

    const int N = in_sizes[0] / 3;
    const int E = in_sizes[1] / 2;
    const int* src = ei;
    const int* dst = ei + E;

    // ---- workspace bump allocator (256B aligned) ----
    char* base = (char*)d_ws;
    size_t off = 0;
    auto walloc = [&](size_t bytes) -> void* {
        void* p = base + off;
        off = (off + bytes + 255) & ~(size_t)255;
        return p;
    };
    float* deg  = (float*)walloc((size_t)N * 4);
    float* w    = (float*)walloc((size_t)E * 4);
    float* h    = (float*)walloc((size_t)N * 128 * 4);   // Tx0 / activations
    float* Tb   = (float*)walloc((size_t)N * 128 * 4);   // Tx1 / Tx3
    float* tmp  = (float*)walloc((size_t)N * 128 * 4);   // prop scratch
    float* gout = (float*)walloc((size_t)N * 128 * 4);   // gemm output
    unsigned short* Abf = (unsigned short*)walloc((size_t)N * 512 * 2);
    unsigned short* Bp  = (unsigned short*)walloc((size_t)512 * 128 * 2);
    float* mean = (float*)walloc(128 * 4);
    float* istd = (float*)walloc(128 * 4);
    float* c1   = (float*)walloc((size_t)N * 3 * 4);
    float* c2   = (float*)walloc((size_t)N * 3 * 4);
    float* c3   = (float*)walloc((size_t)N * 3 * 4);
    float* t3   = (float*)walloc((size_t)N * 3 * 4);

    const int n128 = N * 128;
    const int n3   = N * 3;
    const dim3 B256(256);
    const int gE    = (E + 255) / 256;
    const int gE8   = (E + 7) / 8;        // wave per edge, 8 waves/block
    const int gN128 = (n128 + 255) / 256;
    const int gN3   = (n3 + 255) / 256;
    const int gGemm = (N + 15) / 16;

    // ---- edge weights ----
    k_zero_f32<<<(N + 255) / 256, B256, 0, stream>>>(deg, N);
    k_degree<<<gE, B256, 0, stream>>>(src, deg, E);
    k_edgew<<<gE, B256, 0, stream>>>(src, dst, deg, w, E);

    // ---- layer 1 (3 -> 128), leaky_relu + BN ----
    k_zero_f32<<<gN3, B256, 0, stream>>>(c1, n3);
    k_prop3<<<gE, B256, 0, stream>>>(x, src, dst, w, c1, E);          // Tx1
    k_zero_f32<<<gN3, B256, 0, stream>>>(t3, n3);
    k_prop3<<<gE, B256, 0, stream>>>(c1, src, dst, w, t3, E);
    k_cheb_next<<<gN3, B256, 0, stream>>>(t3, x, c2, n3);             // Tx2
    k_zero_f32<<<gN3, B256, 0, stream>>>(t3, n3);
    k_prop3<<<gE, B256, 0, stream>>>(c2, src, dst, w, t3, E);
    k_cheb_next<<<gN3, B256, 0, stream>>>(t3, c1, c3, n3);            // Tx3
    k_gemm_l1<<<gN128, B256, 0, stream>>>(x, c1, c2, c3, W1, b1, gout, n128);
    k_bn_stats<<<128, B256, 0, stream>>>(gout, mean, istd, N);
    k_bn_apply<<<gN128, B256, 0, stream>>>(gout, mean, istd, g1, be1, h, n128);

    // ---- layers 2-4 (128 -> 128) ----
    const float* Ws[3]  = { W2, W3, W4 };
    const float* bs[3]  = { b2, b3, b4 };
    const float* gs[3]  = { g2, g3, nullptr };
    const float* bes[3] = { be2, be3, nullptr };
    const int    acts[3] = { 1, 2, 0 };   // leaky, relu, none

    for (int L = 0; L < 3; ++L) {
        // Tx0 = h
        k_cvt_bf16<<<gN128, B256, 0, stream>>>(h, Abf, 0, n128);
        // Tx1 = prop(Tx0)
        k_zero_f32<<<gN128, B256, 0, stream>>>(Tb, n128);
        k_prop128<<<gE8, B256, 0, stream>>>(h, src, dst, w, Tb, E);
        k_cvt_bf16<<<gN128, B256, 0, stream>>>(Tb, Abf, 128, n128);
        // Tx2 = 2*prop(Tx1) - Tx0   (overwrites h)
        k_zero_f32<<<gN128, B256, 0, stream>>>(tmp, n128);
        k_prop128<<<gE8, B256, 0, stream>>>(Tb, src, dst, w, tmp, E);
        k_cheb_next<<<gN128, B256, 0, stream>>>(tmp, h, h, n128);
        k_cvt_bf16<<<gN128, B256, 0, stream>>>(h, Abf, 256, n128);
        // Tx3 = 2*prop(Tx2) - Tx1   (overwrites Tb)
        k_zero_f32<<<gN128, B256, 0, stream>>>(tmp, n128);
        k_prop128<<<gE8, B256, 0, stream>>>(h, src, dst, w, tmp, E);
        k_cheb_next<<<gN128, B256, 0, stream>>>(tmp, Tb, Tb, n128);
        k_cvt_bf16<<<gN128, B256, 0, stream>>>(Tb, Abf, 384, n128);
        // dense GEMM via WMMA
        k_pack_w<<<65536 / 256, B256, 0, stream>>>(Ws[L], Bp);
        k_gemm_wmma<<<gGemm, B256, 0, stream>>>(Abf, Bp, bs[L], gout, N, acts[L]);
        if (L < 2) {
            k_bn_stats<<<128, B256, 0, stream>>>(gout, mean, istd, N);
            k_bn_apply<<<gN128, B256, 0, stream>>>(gout, mean, istd, gs[L], bes[L], h, n128);
        }
    }

    // ---- final row-wise L2 normalize -> d_out ----
    k_l2norm<<<N, dim3(128), 0, stream>>>(gout, (float*)d_out);
}
